// Decoder_35527969472565
// MI455X (gfx1250) — matrix operand order repaired and checked by reading, not compiled, not run on hardware
//
#include <hip/hip_runtime.h>
#include <hip/hip_bf16.h>
#include <math.h>

// ---------- problem constants ----------
#define B_    32
#define SENC  200
#define T_    400
#define M_    80
#define E2_   512
#define H_    1024
#define IN0_  (M_ + E2_)        // 592
#define K0_   1632              // 592 + 1024, padded to multiple of 32
#define K1_   2048              // 1024 + 1024

// ---------- CDNA5 WMMA types ----------
typedef __attribute__((ext_vector_type(16))) __bf16       v16bf;
typedef __attribute__((ext_vector_type(8)))  float        v8f;
typedef __attribute__((ext_vector_type(4)))  unsigned int v4u;

union FragBF { v16bf v; v4u q[2]; };

__device__ __forceinline__ float sigmoidf_(float x) {
    return 1.0f / (1.0f + __expf(-x));
}

// ======================================================================
// Preamble kernels
// ======================================================================

// Wc0[j][k] = bf16( k<592 ? Wih0[j][k] : k<1616 ? Whh0[j][k-592] : 0 )
__global__ void conv_w0_kernel(const float* __restrict__ Wih0,
                               const float* __restrict__ Whh0,
                               __bf16* __restrict__ Wc0) {
    int j = blockIdx.x;                       // 0..4095
    for (int k = threadIdx.x; k < K0_; k += blockDim.x) {
        float v;
        if (k < IN0_)        v = Wih0[(size_t)j * IN0_ + k];
        else if (k < IN0_ + H_) v = Whh0[(size_t)j * H_ + (k - IN0_)];
        else                 v = 0.0f;
        Wc0[(size_t)j * K0_ + k] = (__bf16)v;
    }
}

// Wc1[j][k] = bf16( k<1024 ? Wih1[j][k] : Whh1[j][k-1024] )
__global__ void conv_w1_kernel(const float* __restrict__ Wih1,
                               const float* __restrict__ Whh1,
                               __bf16* __restrict__ Wc1) {
    int j = blockIdx.x;
    for (int k = threadIdx.x; k < K1_; k += blockDim.x) {
        float v = (k < H_) ? Wih1[(size_t)j * H_ + k]
                           : Whh1[(size_t)j * H_ + (k - H_)];
        Wc1[(size_t)j * K1_ + k] = (__bf16)v;
    }
}

// escore[b][s] = sum_e tanh(enc[b][s][e]) * Wa[H + e]   (step-invariant)
__global__ void escore_kernel(const float* __restrict__ enc,
                              const float* __restrict__ Wa,
                              float* __restrict__ escore) {
    __shared__ float red[256];
    int s = blockIdx.x, b = blockIdx.y, tid = threadIdx.x;
    const float* e = enc + ((size_t)b * SENC + s) * E2_;
    float p = 0.0f;
    for (int i = tid; i < E2_; i += 256) p += tanhf(e[i]) * Wa[H_ + i];
    red[tid] = p; __syncthreads();
    for (int off = 128; off > 0; off >>= 1) {
        if (tid < off) red[tid] += red[tid + off];
        __syncthreads();
    }
    if (tid == 0) escore[b * SENC + s] = red[0];
}

// ======================================================================
// Per-step kernel A: pred_{t-1} + attention context, writes xcat0[:,0:592]
// grid = 32 (one WG per batch row), block = 256
// ======================================================================
__global__ void attn_pred_kernel(const float* __restrict__ enc,
                                 const float* __restrict__ audio,
                                 const float* __restrict__ escore,
                                 const float* __restrict__ Wa,
                                 const float* __restrict__ ba,
                                 const float* __restrict__ Wfc,
                                 const float* __restrict__ bfc,
                                 const float* __restrict__ h1f,
                                 __bf16* __restrict__ xcat0,
                                 float* __restrict__ out,
                                 int t, int doAttn, int doPred) {
    __shared__ float h1s[H_];
    __shared__ float red[256];
    __shared__ float wts[256];
    const int tid = threadIdx.x;
    const int b   = blockIdx.x;

    for (int j = tid; j < H_; j += 256) h1s[j] = h1f[(size_t)b * H_ + j];
    __syncthreads();

    if (doPred) {
        if (tid < M_) {
            const float* wr = Wfc + (size_t)tid * H_;
            float acc = bfc[tid];
            for (int j = 0; j < H_; ++j) acc += h1s[j] * wr[j];
            out[((size_t)b * T_ + (t - 1)) * M_ + tid] = acc;
        }
    }

    if (doAttn) {
        if (t == 0) {
            if (tid < SENC) wts[tid] = 1.0f / (float)SENC;   // mean context
        } else {
            // hdot = tanh(h1) . wa_h  + ba
            float p = 0.0f;
            for (int j = tid; j < H_; j += 256) p += tanhf(h1s[j]) * Wa[j];
            red[tid] = p; __syncthreads();
            for (int off = 128; off > 0; off >>= 1) {
                if (tid < off) red[tid] += red[tid + off];
                __syncthreads();
            }
            float hdot = red[0] + ba[0];
            __syncthreads();
            // softmax over 200 scores
            float sc = (tid < SENC) ? (escore[b * SENC + tid] + hdot) : -3.0e38f;
            red[tid] = sc; __syncthreads();
            for (int off = 128; off > 0; off >>= 1) {
                if (tid < off) red[tid] = fmaxf(red[tid], red[tid + off]);
                __syncthreads();
            }
            float smax = red[0]; __syncthreads();
            float ev = (tid < SENC) ? __expf(sc - smax) : 0.0f;
            red[tid] = ev; __syncthreads();
            for (int off = 128; off > 0; off >>= 1) {
                if (tid < off) red[tid] += red[tid + off];
                __syncthreads();
            }
            float ssum = red[0]; __syncthreads();
            if (tid < SENC) wts[tid] = ev / ssum;
        }
        __syncthreads();

        // ctx[e] = sum_s wts[s] * enc[b][s][e], each thread owns e=tid, tid+256
        float a0 = 0.0f, a1 = 0.0f;
        const float* eb = enc + (size_t)b * SENC * E2_;
        for (int s = 0; s < SENC; ++s) {
            float w = wts[s];
            a0 += w * eb[(size_t)s * E2_ + tid];
            a1 += w * eb[(size_t)s * E2_ + 256 + tid];
        }
        __bf16* xr = xcat0 + (size_t)b * K0_;
        xr[M_ + tid]       = (__bf16)a0;
        xr[M_ + 256 + tid] = (__bf16)a1;
        if (tid < M_)
            xr[tid] = (__bf16)audio[((size_t)b * T_ + t) * M_ + tid];
    }
}

// ======================================================================
// Fused LSTM layer: G = xcat @ Wc^T (+biases), cell update, h/c writeback
// grid = 64 (16 columns each), block = 256 (8 wave32s: wave = gate*2+mtile)
// ======================================================================
template <int K>
__global__ void lstm_kernel(const __bf16* __restrict__ Wc,    // [4096][K]
                            const __bf16* __restrict__ xcat,  // [32][K]
                            const float* __restrict__ bih,
                            const float* __restrict__ bhh,
                            float* __restrict__ c,            // [32][1024]
                            __bf16* __restrict__ hdstA, int strideA, int offA,
                            __bf16* __restrict__ hdstB, int strideB, int offB,
                            float* __restrict__ hf32) {
    __shared__ float g4[4][32][16];

    const int tid   = threadIdx.x;
    const int wave  = tid >> 5;
    const int lane  = tid & 31;
    const int gate  = wave >> 1;        // 0..3 : i,f,g,o
    const int mtile = wave & 1;         // batch rows 0-15 / 16-31
    const int jc    = blockIdx.x * 16;  // column block within 1024
    const int l16   = lane & 15;
    const int half  = lane >> 4;

    // A fragment source row (batch), B fragment source row (weight)
    const __bf16* arow = xcat + (size_t)(mtile * 16 + l16) * K;
    const __bf16* brow = Wc + (size_t)(gate * H_ + jc + l16) * K;

    v8f acc = {};
#pragma unroll 4
    for (int kc = 0; kc < K; kc += 32) {
        FragBF a, bfr;
        // A 16x32 bf16: lane holds k = 8*half+0..7 and 16+8*half+0..7
        a.q[0]   = *(const v4u*)(arow + kc + 8 * half);
        a.q[1]   = *(const v4u*)(arow + kc + 16 + 8 * half);
        // B 32x16 bf16: lane (n=l16) holds k = 8*half+0..7 and 16+8*half+0..7
        bfr.q[0] = *(const v4u*)(brow + kc + 8 * half);
        bfr.q[1] = *(const v4u*)(brow + kc + 16 + 8 * half);
        acc = __builtin_amdgcn_wmma_f32_16x16x32_bf16(
                  false, a.v, false, bfr.v, (short)0, acc, false, false);
    }

    // D(m,n): m = r + 8*half (per-VGPR), n = l16
#pragma unroll
    for (int r = 0; r < 8; ++r)
        g4[gate][mtile * 16 + 8 * half + r][l16] = acc[r];
    __syncthreads();

    // cell update: 512 (b,n) elements by 256 threads
    for (int idx = tid; idx < 512; idx += 256) {
        int bb = idx >> 4, n = idx & 15;
        int j  = jc + n;
        float iv = g4[0][bb][n] + bih[j]           + bhh[j];
        float fv = g4[1][bb][n] + bih[H_ + j]      + bhh[H_ + j];
        float gv = g4[2][bb][n] + bih[2 * H_ + j]  + bhh[2 * H_ + j];
        float ov = g4[3][bb][n] + bih[3 * H_ + j]  + bhh[3 * H_ + j];
        float cprev = c[(size_t)bb * H_ + j];
        float cn = sigmoidf_(fv) * cprev + sigmoidf_(iv) * tanhf(gv);
        float hn = sigmoidf_(ov) * tanhf(cn);
        c[(size_t)bb * H_ + j] = cn;
        __bf16 hb = (__bf16)hn;
        hdstA[(size_t)bb * strideA + offA + j] = hb;
        if (hdstB) hdstB[(size_t)bb * strideB + offB + j] = hb;
        if (hf32)  hf32[(size_t)bb * H_ + j] = hn;
    }
}

// ======================================================================
// Host side
// ======================================================================
static inline char* carve(char*& p, size_t bytes) {
    char* r = p;
    p += (bytes + 255) & ~(size_t)255;
    return r;
}

extern "C" void kernel_launch(void* const* d_in, const int* in_sizes, int n_in,
                              void* d_out, int out_size, void* d_ws, size_t ws_size,
                              hipStream_t stream) {
    (void)in_sizes; (void)n_in; (void)out_size; (void)ws_size;

    const float* enc   = (const float*)d_in[0];   // [32][200][512]
    const float* audio = (const float*)d_in[1];   // [32][400][80]
    const float* Wih0  = (const float*)d_in[2];   // [4096][592]
    const float* Whh0  = (const float*)d_in[3];   // [4096][1024]
    const float* bih0  = (const float*)d_in[4];
    const float* bhh0  = (const float*)d_in[5];
    const float* Wih1  = (const float*)d_in[6];   // [4096][1024]
    const float* Whh1  = (const float*)d_in[7];   // [4096][1024]
    const float* bih1  = (const float*)d_in[8];
    const float* bhh1  = (const float*)d_in[9];
    const float* Wa    = (const float*)d_in[10];  // [1][1536]
    const float* ba    = (const float*)d_in[11];
    const float* Wfc   = (const float*)d_in[12];  // [80][1024]
    const float* bfc   = (const float*)d_in[13];
    float* out = (float*)d_out;                   // [32][400][80]

    char* p = (char*)d_ws;
    __bf16* Wc0    = (__bf16*)carve(p, (size_t)4096 * K0_ * 2);
    __bf16* Wc1    = (__bf16*)carve(p, (size_t)4096 * K1_ * 2);
    __bf16* xcat0  = (__bf16*)carve(p, (size_t)B_ * K0_ * 2);
    __bf16* xcat1  = (__bf16*)carve(p, (size_t)B_ * K1_ * 2);
    float*  c0     = (float*)carve(p, (size_t)B_ * H_ * 4);
    float*  c1     = (float*)carve(p, (size_t)B_ * H_ * 4);
    float*  h1f    = (float*)carve(p, (size_t)B_ * H_ * 4);
    float*  escore = (float*)carve(p, (size_t)B_ * SENC * 4);

    // ---- preamble (runs every call; deterministic, capture-safe) ----
    conv_w0_kernel<<<4096, 256, 0, stream>>>(Wih0, Whh0, Wc0);
    conv_w1_kernel<<<4096, 256, 0, stream>>>(Wih1, Whh1, Wc1);
    hipMemsetAsync(xcat0, 0, (size_t)B_ * K0_ * 2, stream);   // zero h0_prev + pad
    hipMemsetAsync(xcat1, 0, (size_t)B_ * K1_ * 2, stream);   // zero h1_prev
    hipMemsetAsync(c0,    0, (size_t)B_ * H_ * 4, stream);
    hipMemsetAsync(c1,    0, (size_t)B_ * H_ * 4, stream);
    hipMemsetAsync(h1f,   0, (size_t)B_ * H_ * 4, stream);
    escore_kernel<<<dim3(SENC, B_), 256, 0, stream>>>(enc, Wa, escore);

    // ---- sequential decode ----
    for (int t = 0; t < T_; ++t) {
        attn_pred_kernel<<<B_, 256, 0, stream>>>(
            enc, audio, escore, Wa, ba, Wfc, bfc, h1f, xcat0, out,
            t, /*doAttn=*/1, /*doPred=*/(t > 0) ? 1 : 0);

        // LSTM0: h0 -> xcat0[:,592:1616] (next step) and xcat1[:,0:1024] (now)
        lstm_kernel<K0_><<<64, 256, 0, stream>>>(
            Wc0, xcat0, bih0, bhh0, c0,
            xcat0, K0_, IN0_, xcat1, K1_, 0, (float*)nullptr);

        // LSTM1: h1 -> xcat1[:,1024:2048] (next step) and h1f (attn/pred)
        lstm_kernel<K1_><<<64, 256, 0, stream>>>(
            Wc1, xcat1, bih1, bhh1, c1,
            xcat1, K1_, H_, (__bf16*)nullptr, 0, 0, h1f);
    }

    // final prediction for t = 399
    attn_pred_kernel<<<B_, 256, 0, stream>>>(
        enc, audio, escore, Wa, ba, Wfc, bfc, h1f, xcat0, out,
        T_, /*doAttn=*/0, /*doPred=*/1);
}
